// VOneBlock_67731634258400
// MI455X (gfx1250) — matrix operand order repaired and checked by reading, not compiled, not run on hardware
//
#include <hip/hip_runtime.h>

// ---------------------------------------------------------------------------
// VOneBlock for MI455X (gfx1250, wave32).
//  Stage 1: Gabor quadrature conv (stride2, 5x5, CIN=1) + nonlinearity.
//           Writes y fp32 (numerator) to d_out and y bf16 into a padded
//           NHWC workspace buffer (32,66,66,256) for WMMA consumption.
//  Stage 2: divisive-normalization 3x3 dense cross-channel conv as an
//           implicit GEMM: y tile staged to LDS once per block via the
//           Tensor Data Mover (tensor_load_to_lds + s_wait_tensorcnt),
//           v_wmma_f32_16x16x32_bf16 with fp32 accum (2 pixel tiles per
//           wave to double A-fragment reuse), then out = y / under.
// Workspace layout:
//   [0, 71368704)            : padded y bf16, (32,66,66,256)
//   [71368704, +1179648)     : gaussian bank bf16, [tap(9)][co(256)][ci(256)]
// ---------------------------------------------------------------------------

typedef __bf16 v8bf  __attribute__((ext_vector_type(8)));
typedef __bf16 v16bf __attribute__((ext_vector_type(16)));
typedef float  v8f   __attribute__((ext_vector_type(8)));
typedef unsigned int u32x4 __attribute__((ext_vector_type(4)));
typedef int i32x4 __attribute__((ext_vector_type(4)));
typedef int i32x8 __attribute__((ext_vector_type(8)));

#define YB_BYTES   71368704u          // 32*66*66*256*2
#define YB_U4      4460544            // YB_BYTES / 16
#define GB_ELEMS   589824             // 9*256*256

#ifndef __has_builtin
#define __has_builtin(x) 0
#endif
#if __has_builtin(__builtin_amdgcn_tensor_load_to_lds)
#define USE_TDM 1
#else
#define USE_TDM 0
#endif

// ---------------- zero fill (padded border must be 0) ----------------------
__global__ __launch_bounds__(256) void fill_zero_kernel(uint4* __restrict__ p, int count) {
    int i = blockIdx.x * 256 + threadIdx.x;
    if (i < count) p[i] = make_uint4(0u, 0u, 0u, 0u);
}

// ---------------- gaussian bank fp32 (co,ci,3,3) -> bf16 [t][co][ci] -------
__global__ __launch_bounds__(256) void gconv_kernel(const float* __restrict__ g,
                                                    __bf16* __restrict__ gb) {
    int i = blockIdx.x * 256 + threadIdx.x;         // < 589824
    if (i >= GB_ELEMS) return;
    int t  = i >> 16;                               // tap 0..8
    int rc = i & 65535;                             // co*256 + ci
    gb[i] = (__bf16)g[(size_t)rc * 9 + t];          // src: ((co*256+ci)*3+ky)*3+kx
}

// ---------------- Gabor conv + nonlinearity --------------------------------
__global__ __launch_bounds__(256) void gabor_kernel(const float* __restrict__ x,
                                                    const float* __restrict__ w0,
                                                    const float* __restrict__ w1,
                                                    float*  __restrict__ y,    // d_out NCHW
                                                    __bf16* __restrict__ yb) { // padded NHWC
    __shared__ float xs[35 * 35];
    const int tid  = threadIdx.x;
    const int bid  = blockIdx.x;            // 512 blocks: n(32) x 16 tiles
    const int n    = bid >> 4;
    const int tile = bid & 15;
    const int oh0  = (tile >> 2) << 4;
    const int ow0  = (tile & 3)  << 4;
    const int ih0  = oh0 * 2 - 2;
    const int iw0  = ow0 * 2 - 2;

    for (int i = tid; i < 35 * 35; i += 256) {
        int ly = i / 35, lx = i % 35;
        int ih = ih0 + ly, iw = iw0 + lx;
        float v = 0.0f;
        if (ih >= 0 && ih < 128 && iw >= 0 && iw < 128)
            v = x[(size_t)n * 16384 + ih * 128 + iw];
        xs[i] = v;
    }
    __syncthreads();

    const int py = tid >> 4, px = tid & 15;
    const int oh = oh0 + py, ow = ow0 + px;
    const int xb = (py * 2) * 35 + px * 2;

    for (int c = 0; c < 256; ++c) {
        const float* wp0 = w0 + c * 25;
        float q0 = 0.0f;
        #pragma unroll
        for (int ky = 0; ky < 5; ++ky)
            #pragma unroll
            for (int kx = 0; kx < 5; ++kx)
                q0 = fmaf(wp0[ky * 5 + kx], xs[xb + ky * 35 + kx], q0);

        float val;
        if (c < 128) {                       // simple cells: relu(q0)
            val = fmaxf(q0, 0.0f);
        } else {                             // complex cells: |q|/sqrt(2)
            const float* wp1 = w1 + c * 25;
            float q1 = 0.0f;
            #pragma unroll
            for (int ky = 0; ky < 5; ++ky)
                #pragma unroll
                for (int kx = 0; kx < 5; ++kx)
                    q1 = fmaf(wp1[ky * 5 + kx], xs[xb + ky * 35 + kx], q1);
            val = sqrtf(q0 * q0 + q1 * q1) * 0.70710678118654752f;
        }
        float yv = 25.0f * val;

        y[(((size_t)n * 256 + c) << 12) + oh * 64 + ow] = yv;
        yb[((((size_t)n * 66 + oh + 1) * 66) + (ow + 1)) * 256 + c] = (__bf16)yv;
    }
}

// ---------------- divisive normalization via WMMA + TDM --------------------
// grid.x = 4096 (32 px per block along one output row), grid.y = 2 co halves.
// 8 waves / block: wave w owns co tile [co0, co0+16) x 32 px (2 accumulators).
// y tile (3 rows x 34 px x 256 ch bf16 = 52224 B) staged once into LDS.
__global__ __launch_bounds__(256) void dn_wmma_kernel(const __bf16* __restrict__ yb,
                                                      const __bf16* __restrict__ gb,
                                                      float* __restrict__ out) {
    __shared__ __bf16 lds_y[3 * 34 * 256];   // [row][col][ci], 52224 bytes

    const int tid   = threadIdx.x;
    const int lane  = tid & 31;
    const int wave  = tid >> 5;
    const int co0   = blockIdx.y * 128 + wave * 16;

    const int P0  = blockIdx.x << 5;        // first pixel of block (32 px)
    const int n   = P0 >> 12;
    const int rem = P0 & 4095;
    const int oh  = rem >> 6;
    const int ow0 = rem & 63;               // 0 or 32

    // ---- stage y tile: padded rows oh..oh+2, padded cols ow0..ow0+33 ------
    const __bf16* gsrc = yb + (((size_t)n * 66 + oh) * 66 + ow0) * 256;
#if USE_TDM
    if (wave == 0) {
        const unsigned long long ga = (unsigned long long)(const void*)gsrc;
        const unsigned ldsAddr = (unsigned)(unsigned long long)(&lds_y[0]);
        // ---- D# group 0: count=1 | lds_addr | global_addr | type=2 --------
        u32x4 g0;
        g0[0] = 1u;                                     // count=1, user mode
        g0[1] = ldsAddr;                                // LDS byte address
        g0[2] = (unsigned)(ga & 0xFFFFFFFFull);         // global_addr[31:0]
        g0[3] = (unsigned)((ga >> 32) & 0x1FFFFFFull)   // global_addr[56:32]
              | (2u << 30);                             // type=2 ("image")
        // ---- D# group 1: sizes/strides (2D tile: 8704 x 3 rows) -----------
        const unsigned TD0  = 66u * 256u;               // tensor_dim0 = 16896
        const unsigned TD1  = 66u * 32u;                // tensor_dim1 (rows total)
        const unsigned TIL0 = 34u * 256u;               // tile_dim0 = 8704
        const unsigned TIL1 = 3u;                       // tile_dim1 = 3 rows
        const unsigned STR0 = 66u * 256u;               // tensor_dim0_stride
        i32x8 g1;
        g1[0] = (int)(1u << 16);                        // data_size=1 -> 2 bytes
        g1[1] = (int)((TD0 & 0xFFFFu) << 16);           // tensor_dim0[15:0]
        g1[2] = (int)((TD0 >> 16) | ((TD1 & 0xFFFFu) << 16));
        g1[3] = (int)((TD1 >> 16) | (TIL0 << 16));      // tile_dim0
        g1[4] = (int)(TIL1);                            // tile_dim1, tile_dim2=0
        g1[5] = (int)STR0;                              // stride0[31:0]
        g1[6] = 0;                                      // stride0[47:32], stride1[15:0]
        g1[7] = 0;                                      // stride1[47:16]
        i32x4 gz = {0, 0, 0, 0};
#if __clang_major__ >= 23
        i32x8 gz8 = {0, 0, 0, 0, 0, 0, 0, 0};
        __builtin_amdgcn_tensor_load_to_lds(g0, g1, gz, gz, gz8, 0);
#else
        __builtin_amdgcn_tensor_load_to_lds(g0, g1, gz, gz, 0);
#endif
        __builtin_amdgcn_s_wait_tensorcnt(0);
    }
    __syncthreads();
#else
    // fallback: cooperative copy, 3 rows x 1088 uint4
    for (int i = tid; i < 3 * 1088; i += 256) {
        int r = i / 1088, e = i - r * 1088;
        const uint4* src = (const uint4*)(yb + (((size_t)n * 66 + oh + r) * 66 + ow0) * 256);
        ((uint4*)lds_y)[r * 1088 + e] = src[e];
    }
    __syncthreads();
#endif

    const int  px = lane & 15;
    const bool hi = lane >= 16;
    const int aRow = co0 + (hi ? lane - 16 : lane);   // A-matrix M row
    const int asub = hi ? 16 : 0;                     // A lane K sub-offset
    const int bsub = hi ? 16 : 0;                     // B lane K sub-offset

    v8f acc0 = {0.0f, 0.0f, 0.0f, 0.0f, 0.0f, 0.0f, 0.0f, 0.0f};
    v8f acc1 = acc0;

    for (int t = 0; t < 9; ++t) {
        const int ky = t / 3, kx = t % 3;
        // LDS col index for pixel (pt*16+px): (ky*34 + pt*16+px + kx)*256
        const __bf16* b0 = &lds_y[(ky * 34 + px + kx) * 256];
        const __bf16* b1 = b0 + 16 * 256;
        const __bf16* abase = gb + ((size_t)t << 16) + (size_t)aRow * 256;
        #pragma unroll
        for (int ks = 0; ks < 8; ++ks) {
            const int k0 = ks << 5;
            v8bf alo = *(const v8bf*)(abase + k0 + asub);
            v8bf ahi = *(const v8bf*)(abase + k0 + asub + 8);
            v16bf a = __builtin_shufflevector(alo, ahi,
                        0, 1, 2, 3, 4, 5, 6, 7, 8, 9, 10, 11, 12, 13, 14, 15);
            v16bf bf0 = *(const v16bf*)(b0 + k0 + bsub);
            v16bf bf1 = *(const v16bf*)(b1 + k0 + bsub);
            acc0 = __builtin_amdgcn_wmma_f32_16x16x32_bf16(
                       false, a, false, bf0, (short)0, acc0, false, false);
            acc1 = __builtin_amdgcn_wmma_f32_16x16x32_bf16(
                       false, a, false, bf1, (short)0, acc1, false, false);
        }
    }

    // D layout: VGPR r -> M=r (lanes 0-15) / M=8+r (lanes 16-31), N = lane%16.
    // out = y / under; numerator y fp32 already sits in d_out.
    #pragma unroll
    for (int pt = 0; pt < 2; ++pt) {
        const v8f acc = pt ? acc1 : acc0;
        const size_t base = (((size_t)n * 256 + co0) << 12)
                          + (size_t)oh * 64 + ow0 + pt * 16 + px;
        #pragma unroll
        for (int r = 0; r < 8; ++r) {
            const int m = hi ? (8 + r) : r;
            const size_t idx = base + ((size_t)m << 12);
            out[idx] = out[idx] / acc[r];
        }
    }
}

// ---------------------------------------------------------------------------
extern "C" void kernel_launch(void* const* d_in, const int* in_sizes, int n_in,
                              void* d_out, int out_size, void* d_ws, size_t ws_size,
                              hipStream_t stream) {
    const float* x  = (const float*)d_in[0];   // (32,1,128,128)
    const float* w0 = (const float*)d_in[1];   // (256,1,5,5)
    const float* w1 = (const float*)d_in[2];   // (256,1,5,5)
    const float* g  = (const float*)d_in[3];   // (256,256,3,3)
    float* out = (float*)d_out;                // (32,256,64,64)

    __bf16* yb = (__bf16*)d_ws;                          // padded NHWC y, bf16
    __bf16* gb = (__bf16*)((char*)d_ws + YB_BYTES);      // [t][co][ci] bf16

    fill_zero_kernel<<<17424, 256, 0, stream>>>((uint4*)yb, YB_U4);
    gconv_kernel<<<2304, 256, 0, stream>>>(g, gb);
    gabor_kernel<<<512, 256, 0, stream>>>(x, w0, w1, out, yb);
    dn_wmma_kernel<<<dim3(4096, 2), 256, 0, stream>>>(yb, gb, out);
}
